// ScaledDotProductAttention_52209622450435
// MI455X (gfx1250) — matrix operand order repaired and checked by reading, not compiled, not run on hardware
//
#include <hip/hip_runtime.h>
#include <hip/hip_bf16.h>

#define HID   512
#define BATCH 16
#define SEQ   4096

typedef __attribute__((ext_vector_type(16))) __bf16 v16bf;
typedef __attribute__((ext_vector_type(8)))  float  v8f;

static __device__ __forceinline__ __bf16 f2bf(float f) { return (__bf16)f; }

// Split-precision (bf16 hi/lo x3) GEMM: D[16][512] = (A[16][512] * B + bias) * outScale.
// TRANSB=1: B[k][n] = W[n*512+k] (i.e. A @ W^T).  TRANSB=0: B[k][n] = W[k*512+n].
// HAS_BIAS templated so the epilogue is branch-free straight-line code.
// A is generic-pointer with row stride aStride. 256 threads = 8 waves, 4 n-tiles each.
// Fragment layouts per CDNA5 ISA 7.12.2 (wave32).
template <bool TRANSB, bool HAS_BIAS>
__device__ void gemm16_512_512(const float* A, long aStride,
                               const float* __restrict__ Bg,
                               const float* __restrict__ bias,
                               float outScale, float* D, long dStride)
{
    const int tid  = threadIdx.x;
    const int lane = tid & 31;
    const int wave = tid >> 5;
    const int half = lane >> 4;   // lane group: 0 = lanes 0-15, 1 = lanes 16-31
    const int mrow = lane & 15;   // A row held by this lane
    const int ncol = lane & 15;   // B/C/D column within the 16-wide tile

    for (int t = 0; t < 4; ++t) {
        const int nBase = (wave * 4 + t) * 16;
        v8f acc = {};
        for (int kk = 0; kk < 16; ++kk) {           // K = 512 in steps of 32
            const int k0 = kk * 32;
            v16bf a_hi, a_lo, b_hi, b_lo;
            // A 16x32 bf16 fragment: elem 2v+p holds K = ((v&4)?16:0) + half*8 + (v&3)*2 + p
            #pragma unroll
            for (int v = 0; v < 8; ++v) {
                #pragma unroll
                for (int p = 0; p < 2; ++p) {
                    const int kl = ((v & 4) ? 16 : 0) + half * 8 + ((v & 3) << 1) + p;
                    const float a = A[(long)mrow * aStride + (k0 + kl)];
                    const __bf16 hi = f2bf(a);
                    a_hi[2 * v + p] = hi;
                    a_lo[2 * v + p] = f2bf(a - (float)hi);
                }
            }
            // B 32x16 bf16 fragment: elem 2v+p holds K = ((v&4)?16:0) + half*8 + (v&3)*2 + p (same as A)
            #pragma unroll
            for (int v = 0; v < 8; ++v) {
                #pragma unroll
                for (int p = 0; p < 2; ++p) {
                    const int k = k0 + ((v & 4) ? 16 : 0) + half * 8 + ((v & 3) << 1) + p;
                    const float b = TRANSB ? Bg[(long)(nBase + ncol) * HID + k]
                                           : Bg[(long)k * HID + (nBase + ncol)];
                    const __bf16 hi = f2bf(b);
                    b_hi[2 * v + p] = hi;
                    b_lo[2 * v + p] = f2bf(b - (float)hi);
                }
            }
            acc = __builtin_amdgcn_wmma_f32_16x16x32_bf16(false, a_hi, false, b_hi,
                                                          (short)0, acc, false, false);
            acc = __builtin_amdgcn_wmma_f32_16x16x32_bf16(false, a_lo, false, b_hi,
                                                          (short)0, acc, false, false);
            acc = __builtin_amdgcn_wmma_f32_16x16x32_bf16(false, a_hi, false, b_lo,
                                                          (short)0, acc, false, false);
        }
        // C/D: VGPR r -> M = r + 8*half, N = nBase + ncol
        const int col = nBase + ncol;
        const float bcol = HAS_BIAS ? bias[col] : 0.0f;
        #pragma unroll
        for (int r = 0; r < 8; ++r) {
            const int m = r + half * 8;
            D[(long)m * dStride + col] = (acc[r] + bcol) * outScale;
        }
    }
}

// K1: q = x[:, SEQ-1, :] @ Wq^T + bq ;  qk = (q @ Wk) / sqrt(HID). Also zeroes xb.
// (bk contributes a per-batch constant to scores -> softmax-invariant -> dropped.)
__global__ void __launch_bounds__(256)
k_prep(const float* __restrict__ x, const float* __restrict__ Wq,
       const float* __restrict__ bq, const float* __restrict__ Wk,
       float* __restrict__ q, float* __restrict__ qk, float* __restrict__ xb)
{
    for (int i = threadIdx.x; i < BATCH * HID; i += 256) xb[i] = 0.0f;
    gemm16_512_512<true, true>(x + (long)(SEQ - 1) * HID, (long)SEQ * HID,
                               Wq, bq, 1.0f, q, HID);
    __threadfence();
    __syncthreads();
    const float rscale = 0.04419417382415922f;   // 1/sqrt(512)
    gemm16_512_512<false, false>(q, HID, Wk, nullptr, rscale, qk, HID);
}

// K2: scores[b,n] = x[b,n,:] . qk[b,:]   (streaming pass 1 over x, one wave per row)
__global__ void __launch_bounds__(256)
k_scores(const float* __restrict__ x, const float* __restrict__ qk,
         float* __restrict__ scores)
{
    __shared__ float qs[HID];
    const int b = blockIdx.x >> 9;               // 512 blocks per batch
    const int nBase = (blockIdx.x & 511) * 8;    // 8 rows (waves) per block
    const int tid = threadIdx.x;
    for (int i = tid; i < HID; i += 256) qs[i] = qk[b * HID + i];
    __syncthreads();
    const int wave = tid >> 5, lane = tid & 31;
    const int n = nBase + wave;
    const float4* xr = (const float4*)(x + ((long)b * SEQ + n) * HID);
    const float4* qr = (const float4*)qs;
    float acc = 0.0f;
    #pragma unroll
    for (int i = 0; i < 4; ++i) {                // 4 coalesced float4 loads per lane
        const float4 xv = xr[i * 32 + lane];
        const float4 qv = qr[i * 32 + lane];
        acc += xv.x * qv.x + xv.y * qv.y + xv.z * qv.z + xv.w * qv.w;
    }
    #pragma unroll
    for (int m = 16; m >= 1; m >>= 1) acc += __shfl_xor(acc, m, 32);
    if (lane == 0) scores[(long)b * SEQ + n] = acc;
}

// K3: in-place softmax over N per batch
__global__ void __launch_bounds__(256)
k_softmax(float* __restrict__ scores)
{
    __shared__ float red[256];
    const int tid = threadIdx.x;
    float* s = scores + (long)blockIdx.x * SEQ;
    float mx = -1e30f;
    for (int i = tid; i < SEQ; i += 256) mx = fmaxf(mx, s[i]);
    red[tid] = mx; __syncthreads();
    for (int st = 128; st >= 1; st >>= 1) {
        if (tid < st) red[tid] = fmaxf(red[tid], red[tid + st]);
        __syncthreads();
    }
    mx = red[0]; __syncthreads();
    float sum = 0.0f;
    for (int i = tid; i < SEQ; i += 256) sum += __expf(s[i] - mx);
    red[tid] = sum; __syncthreads();
    for (int st = 128; st >= 1; st >>= 1) {
        if (tid < st) red[tid] += red[tid + st];
        __syncthreads();
    }
    const float inv = 1.0f / red[0];
    for (int i = tid; i < SEQ; i += 256) s[i] = __expf(s[i] - mx) * inv;
}

// K4: xb[b,:] += sum_n w[b,n] * x[b,n,:]   (pass 2 over x, largely L2-resident)
__global__ void __launch_bounds__(256)
k_accum(const float* __restrict__ x, const float* __restrict__ w,
        float* __restrict__ xb)
{
    __shared__ float wl[128];
    const int b = blockIdx.x >> 5;
    const int n0 = (blockIdx.x & 31) * 128;
    const int tid = threadIdx.x;
    if (tid < 128) wl[tid] = w[(long)b * SEQ + n0 + tid];
    __syncthreads();
    float a0 = 0.0f, a1 = 0.0f;
    const int h = tid * 2;
    for (int r = 0; r < 128; ++r) {
        const float2 xv = *(const float2*)(x + ((long)b * SEQ + n0 + r) * HID + h);
        const float wt = wl[r];
        a0 += wt * xv.x;
        a1 += wt * xv.y;
    }
    atomicAdd(&xb[b * HID + h], a0);
    atomicAdd(&xb[b * HID + h + 1], a1);
}

// K5: out = xb @ Wv^T + bv
__global__ void __launch_bounds__(256)
k_out(const float* __restrict__ xb, const float* __restrict__ Wv,
      const float* __restrict__ bv, float* __restrict__ out)
{
    gemm16_512_512<true, true>(xb, HID, Wv, bv, 1.0f, out, HID);
}

extern "C" void kernel_launch(void* const* d_in, const int* in_sizes, int n_in,
                              void* d_out, int out_size, void* d_ws, size_t ws_size,
                              hipStream_t stream)
{
    const float* x  = (const float*)d_in[0];
    const float* Wq = (const float*)d_in[1];
    const float* bq = (const float*)d_in[2];
    const float* Wk = (const float*)d_in[3];
    // d_in[4] = bk: softmax shift-invariant, drops out analytically
    const float* Wv = (const float*)d_in[5];
    const float* bv = (const float*)d_in[6];
    float* out = (float*)d_out;

    float* ws     = (float*)d_ws;
    float* qk     = ws;                        // 16*512
    float* q      = ws + 1 * BATCH * HID;      // 16*512
    float* xb     = ws + 2 * BATCH * HID;      // 16*512
    float* scores = ws + 3 * BATCH * HID;      // 16*4096  (becomes weights in-place)

    k_prep   <<<dim3(1),               dim3(256), 0, stream>>>(x, Wq, bq, Wk, q, qk, xb);
    k_scores <<<dim3(BATCH * SEQ / 8), dim3(256), 0, stream>>>(x, qk, scores);
    k_softmax<<<dim3(BATCH),           dim3(256), 0, stream>>>(scores);
    k_accum  <<<dim3(BATCH * 32),      dim3(256), 0, stream>>>(x, scores, xb);
    k_out    <<<dim3(1),               dim3(256), 0, stream>>>(xb, Wv, bv, out);
}
